// MaskDPAttention_15788299780316
// MI455X (gfx1250) — matrix operand order repaired and checked by reading, not compiled, not run on hardware
//
#include <hip/hip_runtime.h>
#include <math.h>

typedef __attribute__((ext_vector_type(16))) _Float16 v16h;
typedef __attribute__((ext_vector_type(8)))  float    v8f;

#define SEQ 2048
#define DD  64
#define QT  16           // query rows per workgroup
#define NW  4            // waves per block (wave32)
#define NT  128          // threads per block
#define SCS 2056         // padded score-row stride (f32): 2056 % 64 = 8 -> conflict-free softmax
#define VTS 33           // padded transposed-V stride (f32): gcd(33,64)=1 -> conflict-free
#define VCH (DD*VTS)     // one V-chunk buffer (f32 elements)

// dynamic LDS layout (bytes):
//   sc      : float[16][SCS]      score/prob strip     131584
//   vtf     : float[2][64][VTS]   V chunks (dbl buf)    16896
//   red     : float[16][8]          partial max/sum       512
//   rowstat : float[16]                                    64
#define SMEM_BYTES (QT*SCS*4 + 2*VCH*4 + 16*8*4 + 16*4)

#if defined(__has_builtin)
#if __has_builtin(__builtin_amdgcn_global_load_async_to_lds_b32) && \
    __has_builtin(__builtin_amdgcn_s_wait_asynccnt)
#define USE_ASYNC 1
#endif
#endif
#ifndef USE_ASYNC
#define USE_ASYNC 0
#endif

typedef __attribute__((address_space(1))) int g_int;
typedef __attribute__((address_space(3))) int l_int;

__device__ __forceinline__ void stage_v_chunk(const float* __restrict__ Vsrc,
                                              float* __restrict__ dstbuf, int tid)
{
    // 32 k-rows x 64 cols, scattered transposed into dstbuf[cc*VTS + kk]
    #pragma unroll
    for (int i = 0; i < 16; ++i) {
        const int idx = i * NT + tid;
        const int kk  = idx >> 6;
        const int cc  = idx & (DD - 1);
#if USE_ASYNC
        __builtin_amdgcn_global_load_async_to_lds_b32(
            (g_int*)(Vsrc + (size_t)kk * DD + cc),
            (l_int*)(dstbuf + cc * VTS + kk),
            0, 0);
#else
        dstbuf[cc * VTS + kk] = Vsrc[(size_t)kk * DD + cc];
#endif
    }
}

extern "C" __global__ __launch_bounds__(NT)
void mdpa_fused_kernel(const float* __restrict__ Q, const float* __restrict__ K,
                       const float* __restrict__ V, const unsigned char* __restrict__ M,
                       float* __restrict__ ATT, float* __restrict__ CTX)
{
    extern __shared__ char smem_raw[];
    float*    sc      = (float*)smem_raw;
    float*    vtf     = (float*)(smem_raw + QT*SCS*4);
    float*    red     = (float*)(smem_raw + QT*SCS*4 + 2*VCH*4);
    float*    rowstat = red + 16*8;

    const int tid  = threadIdx.x;
    const int lane = tid & 31;
    const int wave = tid >> 5;
    const int r    = lane & 15;      // A-operand row / B-operand column
    const int half = lane >> 4;

    const int bh = blockIdx.x >> 7;            // SEQ/QT = 128 q-tiles per (b,h)
    const int q0 = (blockIdx.x & 127) * QT;

    const size_t qkv_base  = (size_t)bh * SEQ * DD;
    const size_t mask_base = ((size_t)bh * SEQ + q0) * SEQ;   // also attention base

    // ---- Q tile -> two A-operands (D chunks 0..31, 32..63), f32 -> f16 ----
    v16h qa0, qa1;
    {
        const float* qrow = Q + qkv_base + (size_t)(q0 + r) * DD;
        #pragma unroll
        for (int e = 0; e < 16; ++e) {
            const int ka = e + half * 16;                      // A-operand K map
            qa0[e] = (_Float16)qrow[ka];
            qa1[e] = (_Float16)qrow[32 + ka];
        }
    }

    // ---- phase 1: scores = (Q K^T)*scale, masked -> LDS strip ----
    for (int j = wave; j < SEQ/16; j += NW) {
        const int n0 = j * 16;
        const float* krow = K + qkv_base + (size_t)(n0 + r) * DD;
        if (j + NW < SEQ/16)   // warm caches for the next key tile
            __builtin_prefetch(K + qkv_base + (size_t)((j + NW)*16 + r) * DD, 0, 0);

        v16h kb0, kb1;
        #pragma unroll
        for (int e = 0; e < 16; ++e) {
            const int kb = e + half * 16;                      // B-operand K map
            kb0[e] = (_Float16)krow[kb];
            kb1[e] = (_Float16)krow[32 + kb];
        }
        v8f c = {};
        c = __builtin_amdgcn_wmma_f32_16x16x32_f16(false, qa0, false, kb0, (short)0, c, false, false);
        c = __builtin_amdgcn_wmma_f32_16x16x32_f16(false, qa1, false, kb1, (short)0, c, false, false);

        #pragma unroll
        for (int i = 0; i < 8; ++i) {
            const int m = i + half * 8;                        // C/D row
            float s = c[i] * 0.125f;                           // 1/sqrt(64)
            // mask is a pure stream (no reuse): keep it out of L2's way
            if (__builtin_nontemporal_load(&M[mask_base + (size_t)m * SEQ + n0 + r]))
                s = -1.0e9f;
            sc[m * SCS + n0 + r] = s;
        }
    }
    __syncthreads();

    // ---- phase 2: softmax over each 2048-wide row, in LDS (conflict-free) ----
    const int rr = tid >> 3;     // 8 threads per row
    const int g  = tid & 7;

    float m0 = -3.0e38f;
    for (int cix = g; cix < SEQ; cix += 8)
        m0 = fmaxf(m0, sc[rr*SCS + cix]);
    red[rr*8 + g] = m0;
    __syncthreads();
    if (g == 0) {
        float mm = red[rr*8];
        #pragma unroll
        for (int t = 1; t < 8; ++t) mm = fmaxf(mm, red[rr*8 + t]);
        rowstat[rr] = mm;
    }
    __syncthreads();

    const float rm = rowstat[rr];
    float s0 = 0.f;
    for (int cix = g; cix < SEQ; cix += 8) {
        const float e = __expf(sc[rr*SCS + cix] - rm);
        sc[rr*SCS + cix] = e;
        s0 += e;
    }
    red[rr*8 + g] = s0;
    __syncthreads();
    if (g == 0) {
        float ss = 0.f;
        #pragma unroll
        for (int t = 0; t < 8; ++t) ss += red[rr*8 + t];
        rowstat[rr] = 1.0f / ss;
    }
    __syncthreads();

    // normalize in LDS + coalesced NON-TEMPORAL stream of the attention strip
    for (int ix = tid; ix < QT*SEQ; ix += NT) {
        const int rw  = ix >> 11;
        const int col = ix & (SEQ - 1);
        const float p = sc[rw*SCS + col] * rowstat[rw];
        sc[rw*SCS + col] = p;
        __builtin_nontemporal_store(p, &ATT[mask_base + (size_t)rw * SEQ + col]);
    }

    // ---- phase 3: context = P @ V, async double-buffered V staging ----
    v8f acc = {};
    const int cbase = wave * 16;               // output column group per wave
    const float* Vbh = V + qkv_base;

    stage_v_chunk(Vbh, vtf, tid);              // prefetch chunk 0 into buffer 0
    for (int ck = 0; ck < SEQ/32; ++ck) {
        if (ck + 1 < SEQ/32)                   // overlap: issue next chunk's copy
            stage_v_chunk(Vbh + (size_t)(ck + 1) * 32 * DD, vtf + ((ck + 1) & 1) * VCH, tid);
#if USE_ASYNC
        if (ck + 1 < SEQ/32) __builtin_amdgcn_s_wait_asynccnt(16);  // chunk ck resident (in-order)
        else                 __builtin_amdgcn_s_wait_asynccnt(0);
#endif
        __syncthreads();                       // all waves' portions of chunk ck visible

        v16h pa, vb;
        const float* prow = sc  + r * SCS + ck * 32;
        const float* vcol = vtf + (ck & 1) * VCH + (cbase + r) * VTS;
        #pragma unroll
        for (int e = 0; e < 16; ++e) {
            const int ka = e + half * 16;                      // A map (P rows)
            const int kb = e + half * 16;                      // B map (V cols)
            pa[e] = (_Float16)prow[ka];
            vb[e] = (_Float16)vcol[kb];
        }
        acc = __builtin_amdgcn_wmma_f32_16x16x32_f16(false, pa, false, vb, (short)0, acc, false, false);
        __syncthreads();                       // buffer (ck&1) free for re-staging
    }

    #pragma unroll
    for (int i = 0; i < 8; ++i) {
        const int m = i + half * 8;
        CTX[((size_t)bh * SEQ + q0 + m) * DD + cbase + r] = acc[i];
    }
}

extern "C" void kernel_launch(void* const* d_in, const int* in_sizes, int n_in,
                              void* d_out, int out_size, void* d_ws, size_t ws_size,
                              hipStream_t stream)
{
    (void)in_sizes; (void)n_in; (void)out_size; (void)d_ws; (void)ws_size;
    const float*         Q = (const float*)d_in[0];
    const float*         K = (const float*)d_in[1];
    const float*         V = (const float*)d_in[2];
    const unsigned char* M = (const unsigned char*)d_in[3];   // jnp.bool_ -> 1 byte

    float* ATT = (float*)d_out;
    float* CTX = ATT + (size_t)2 * 16 * SEQ * SEQ;            // outputs concat flat

    // ~146 KB dynamic LDS (two blocks still fit in the 320 KB WGP LDS)
    (void)hipFuncSetAttribute(reinterpret_cast<const void*>(mdpa_fused_kernel),
                              hipFuncAttributeMaxDynamicSharedMemorySize,
                              (int)SMEM_BYTES);

    const dim3 grid(2 * 16 * (SEQ / QT));   // B*H*(S/16) = 4096 workgroups
    mdpa_fused_kernel<<<grid, NT, SMEM_BYTES, stream>>>(Q, K, V, M, ATT, CTX);
}
